// CustomW8A8BFP32OFP32Linear_30709016166655
// MI455X (gfx1250) — 8-bit matrix operand order repaired and checked by reading, compiled, not run on hardware
//
#include <hip/hip_runtime.h>
#include <cstdint>


typedef __attribute__((ext_vector_type(8))) int      v8i;
typedef __attribute__((ext_vector_type(4))) int      v4i;
typedef __attribute__((ext_vector_type(8))) _Float16 v8h;

#define QMAXF 127.0f

__device__ __forceinline__ void async_b128_to_lds(uint32_t lds_off, uint64_t gaddr) {
  asm volatile("global_load_async_to_lds_b128 %0, %1, off"
               :: "v"(lds_off), "v"(gaddr) : "memory");
}
__device__ __forceinline__ void wait_async_lds() {
  asm volatile("s_wait_asynccnt 0x0" ::: "memory");
}

__global__ __launch_bounds__(256) void quant_rows(const _Float16* __restrict__ x,
                                                  int8_t* __restrict__ q,
                                                  float* __restrict__ xscale,
                                                  int K) {
  const int token = blockIdx.x;
  const int tid   = threadIdx.x;
  const _Float16* row = x + (size_t)token * K;

  v8h v0 = *(const v8h*)(row + tid * 16);
  v8h v1 = *(const v8h*)(row + tid * 16 + 8);

  float amax = 0.0f;
#pragma unroll
  for (int i = 0; i < 8; ++i) {
    amax = fmaxf(amax, fabsf((float)v0[i]));
    amax = fmaxf(amax, fabsf((float)v1[i]));
  }

  __shared__ float red[256];
  red[tid] = amax;
  __syncthreads();
#pragma unroll
  for (int s = 128; s > 0; s >>= 1) {
    if (tid < s) red[tid] = fmaxf(red[tid], red[tid + s]);
    __syncthreads();
  }
  const float scale = fmaxf(red[0] / QMAXF, 1e-8f);
  if (tid == 0) xscale[token] = scale;
  const float inv = 1.0f / scale;

  int packed[4];
#pragma unroll
  for (int p = 0; p < 4; ++p) {
    int d = 0;
#pragma unroll
    for (int j = 0; j < 4; ++j) {
      const int idx = p * 4 + j;
      const float f = (idx < 8) ? (float)v0[idx] : (float)v1[idx - 8];
      int t = __float2int_rn(f * inv);
      t = (t < -127) ? -127 : (t > 127 ? 127 : t);
      d |= (t & 0xff) << (8 * j);
    }
    packed[p] = d;
  }
  *(v4i*)(q + (size_t)token * K + tid * 16) = *(const v4i*)packed;
}

#define TM 128
#define TN 128
#define TK 64
#define TKP 80
#define ABUF (TM * TKP)
#define BUFSZ ((TM + TN) * TKP)

__global__ __launch_bounds__(256) void w8a8_wmma_gemm(
    const int8_t*  __restrict__ qa,
    const int8_t*  __restrict__ wb,
    const float*   __restrict__ xscale,
    const _Float16* __restrict__ wscale,
    _Float16* __restrict__ out,
    int M, int N, int K) {
  __shared__ int8_t smem[2 * BUFSZ];

  const int tid  = threadIdx.x;
  const int lane = tid & 31;
  const int wave = tid >> 5;
  const int half = lane >> 4;
  const int l16  = lane & 15;

  const int m_blk = blockIdx.y * TM;
  const int n_blk = blockIdx.x * TN;
  const int wm = (wave & 1) * 64;
  const int wn = (wave >> 1) * 32;

  const v8i vzero = {0, 0, 0, 0, 0, 0, 0, 0};
  v8i acc[4][2];
#pragma unroll
  for (int mi = 0; mi < 4; ++mi)
#pragma unroll
    for (int ni = 0; ni < 2; ++ni) acc[mi][ni] = vzero;

  const int srow = tid >> 2;
  const int scol = (tid & 3) * 16;
  const uint32_t lds_base = (uint32_t)(uintptr_t)&smem[0];
  const uint64_t gA = (uint64_t)(uintptr_t)(qa + (size_t)(m_blk + srow) * K + scol);
  const uint64_t gB = (uint64_t)(uintptr_t)(wb + (size_t)(n_blk + srow) * K + scol);
  const uint64_t rowskip = (uint64_t)64 * (uint64_t)K;

  auto stage = [&](int buf, int k0) {
    const uint32_t a0 = lds_base + (uint32_t)buf * BUFSZ + (uint32_t)(srow * TKP + scol);
    const uint32_t b0 = a0 + ABUF;
    async_b128_to_lds(a0,            gA + (uint64_t)k0);
    async_b128_to_lds(a0 + 64 * TKP, gA + rowskip + (uint64_t)k0);
    async_b128_to_lds(b0,            gB + (uint64_t)k0);
    async_b128_to_lds(b0 + 64 * TKP, gB + rowskip + (uint64_t)k0);
  };

  auto kstep = [&](int buf, int k0, bool prefetch_next) {
    if (prefetch_next) stage(buf ^ 1, k0 + TK);

    const int8_t* sA = &smem[buf * BUFSZ];
    const int8_t* sB = &smem[buf * BUFSZ + ABUF];

    v8i bf[2], af[4];
#pragma unroll
    for (int ni = 0; ni < 2; ++ni) {
      const int* pb = (const int*)(sB + (wn + ni * 16 + l16) * TKP);
#pragma unroll
      for (int j = 0; j < 4; ++j) {
        bf[ni][j]     = pb[half * 4 + j];
        bf[ni][4 + j] = pb[8 + half * 4 + j];
      }
    }
#pragma unroll
    for (int mi = 0; mi < 4; ++mi) {
      const int* pa = (const int*)(sA + (wm + mi * 16 + l16) * TKP);
#pragma unroll
      for (int j = 0; j < 4; ++j) {
        af[mi][j]     = pa[half * 4 + j];
        af[mi][4 + j] = pa[8 + half * 4 + j];
      }
    }
#pragma unroll
    for (int mi = 0; mi < 4; ++mi)
#pragma unroll
      for (int ni = 0; ni < 2; ++ni)
        acc[mi][ni] = __builtin_amdgcn_wmma_i32_16x16x64_iu8(
            true, af[mi], true, bf[ni], acc[mi][ni], false, false);

    wait_async_lds();
    __syncthreads();
  };

  stage(0, 0);
  wait_async_lds();
  __syncthreads();

  for (int k0 = 0; k0 < K; k0 += 2 * TK) {
    kstep(0, k0, true);
    kstep(1, k0 + TK, k0 + 2 * TK < K);
  }

#pragma unroll
  for (int ni = 0; ni < 2; ++ni) {
    const int n = n_blk + wn + ni * 16 + l16;
    const float wsc = (float)wscale[n];
#pragma unroll
    for (int mi = 0; mi < 4; ++mi) {
      const int mrow = m_blk + wm + mi * 16 + half * 8;
#pragma unroll
      for (int r = 0; r < 8; ++r) {
        const int m = mrow + r;
        const float v = (float)acc[mi][ni][r] * xscale[m] * wsc;
        out[(size_t)m * N + n] = (_Float16)v;
      }
    }
  }
}

extern "C" void kernel_launch(void* const* d_in, const int* in_sizes, int n_in,
                              void* d_out, int out_size, void* d_ws, size_t ws_size,
                              hipStream_t stream) {
  const _Float16* x   = (const _Float16*)d_in[0];
  const int8_t*   w   = (const int8_t*)d_in[1];
  const _Float16* wsc = (const _Float16*)d_in[2];
  _Float16* out = (_Float16*)d_out;

  const int K = 4096;
  const int N = in_sizes[2];
  const int M = in_sizes[0] / K;

  int8_t* q  = (int8_t*)d_ws;
  float*  xs = (float*)((char*)d_ws + (size_t)M * K);

  quant_rows<<<M, 256, 0, stream>>>(x, q, xs, K);

  dim3 grid(N / TN, M / TM);
  w8a8_wmma_gemm<<<grid, 256, 0, stream>>>(q, w, xs, wsc, out, M, N, K);
}
